// Head_50096498541091
// MI455X (gfx1250) — matrix operand order repaired and checked by reading, not compiled, not run on hardware
//
#include <hip/hip_runtime.h>
#include <hip/hip_bf16.h>

typedef __attribute__((ext_vector_type(16))) _Float16 v16h;
typedef __attribute__((ext_vector_type(8)))  _Float16 v8h;
typedef __attribute__((ext_vector_type(8)))  float    v8f;
typedef __attribute__((ext_vector_type(4)))  float    v4f;
typedef __attribute__((ext_vector_type(4)))  unsigned int v4u;
typedef __attribute__((ext_vector_type(8)))  int      v8i;
typedef __attribute__((ext_vector_type(4)))  int      v4i;

#define NEMBD 1024
#define HS    64
#define BK    32   // key tile = WMMA K for the PV GEMM
#define SWT   40   // padded stride (halves) for transposed LDS tiles (80B: 16B-aligned runs, distinct banks)

// --- CDNA5 async engine: per-lane 16B global -> LDS copy (ASYNCcnt) ---------
__device__ __forceinline__ void async_g2l_b128(unsigned lds_byte_off,
                                               const void* gptr) {
    asm volatile("global_load_async_to_lds_b128 %0, %1, off"
                 :: "v"(lds_byte_off), "v"((unsigned long long)gptr)
                 : "memory");
}
__device__ __forceinline__ void wait_async0() {
    asm volatile("s_wait_asynccnt 0" ::: "memory");
}

// --- CDNA5 Tensor Data Mover: 1-D tile DMA global -> LDS (TENSORcnt) --------
// D# per CDNA5 ISA.
__device__ __forceinline__ void tdm_load_1d(unsigned lds_byte_off,
                                            const void* gptr,
                                            unsigned nelem /* 2B elements */) {
    unsigned long long ga = (unsigned long long)gptr;
    v4u g0;
    g0[0] = 1u;                                     // count=1, user mode
    g0[1] = lds_byte_off;                           // lds_addr
    g0[2] = (unsigned)(ga & 0xffffffffu);           // global_addr[31:0]
    g0[3] = (unsigned)((ga >> 32) & 0x01ffffffu)    // global_addr[56:32]
          | (2u << 30);                             // type=2 ("image")
    v8i g1 = {};
    g1[0] = (1 << 16);                              // data_size = 1 -> 2 bytes
    g1[1] = (int)((nelem & 0xffffu) << 16);         // tensor_dim0[15:0]
    g1[2] = (int)((nelem >> 16) & 0xffffu)          // tensor_dim0[31:16]
          | (1 << 16);                              // tensor_dim1 = 1
    g1[3] = (int)(nelem << 16);                     // tile_dim0 = nelem
    g1[4] = 1;                                      // tile_dim1 = 1
    g1[5] = (int)nelem;                             // tensor_dim0_stride
    v4i gz = {};
#if defined(__clang_major__) && __clang_major__ >= 23
    v8i gz8 = {};
    __builtin_amdgcn_tensor_load_to_lds(g0, g1, gz, gz, gz8, 0);
#else
    __builtin_amdgcn_tensor_load_to_lds(g0, g1, gz, gz, 0);
#endif
}

// ---------------------------------------------------------------------------
// Kernel 1: q/k/v = x @ W{q,k,v}. 4 waves/block; W chunk staged TRANSPOSED in
// LDS so every B-fragment is two ds_load_b128's.
// ---------------------------------------------------------------------------
__global__ __launch_bounds__(128) void qkv_proj_kernel(
    const float* __restrict__ x,
    const float* __restrict__ Wq, const float* __restrict__ Wk,
    const float* __restrict__ Wv,
    _Float16* __restrict__ qkv, int BT)
{
    __shared__ __align__(16) _Float16 sWt[HS * SWT];    // W chunk, [col][k] (5KB)
    __shared__ __align__(16) _Float16 sOut[4][16 * HS]; // per-wave out tile (8KB)

    const int tid  = threadIdx.x;
    const int wave = tid >> 5;
    const int lane = tid & 31;
    const int tile = blockIdx.x * 4 + wave;   // 16-row tile of x
    const int mat  = blockIdx.y;              // 0=q, 1=k, 2=v
    const float* __restrict__ W = (mat == 0) ? Wq : (mat == 1) ? Wk : Wv;
    _Float16* __restrict__ out = qkv + (size_t)mat * (size_t)BT * HS;

    const bool hi    = lane >= 16;
    const int  aoff0 = hi ? 16 : 0;
    const int  aoff1 = aoff0 + 8;
    const int  koff  = hi ? 16 : 0;

    // staging role: thread t handles column (t&63), k-rows (t>>6)*16..+15
    const int scol = tid & 63;
    const int srh  = tid >> 6;

    v8f acc[4] = {};
    const float* __restrict__ xrow =
        x + (size_t)(tile * 16 + (lane & 15)) * NEMBD;

    for (int kc = 0; kc < NEMBD / 32; ++kc) {
        const int kbase = kc * 32;

        // ---- stage W chunk transposed: coalesced column reads, b128 writes --
        __syncthreads();
        {
            v8h h0, h1;
            #pragma unroll
            for (int i = 0; i < 8; ++i) {
                h0[i] = (_Float16)W[(size_t)(kbase + srh * 16 + i)     * HS + scol];
                h1[i] = (_Float16)W[(size_t)(kbase + srh * 16 + 8 + i) * HS + scol];
            }
            *(v8h*)&sWt[scol * SWT + srh * 16]     = h0;
            *(v8h*)&sWt[scol * SWT + srh * 16 + 8] = h1;
        }
        __syncthreads();

        __builtin_prefetch(xrow + kbase + 32, 0, 1);   // global_prefetch_b8

        // ---- A fragment: four 16B loads (16 contiguous floats) of x, f32 -> f16 ----
        v4f x0 = *(const v4f*)(xrow + kbase + aoff0);
        v4f x1 = *(const v4f*)(xrow + kbase + aoff0 + 4);
        v4f x2 = *(const v4f*)(xrow + kbase + aoff1);
        v4f x3 = *(const v4f*)(xrow + kbase + aoff1 + 4);
        v16h a;
        #pragma unroll
        for (int i = 0; i < 4; ++i) {
            a[i]      = (_Float16)x0[i];
            a[4 + i]  = (_Float16)x1[i];
            a[8 + i]  = (_Float16)x2[i];
            a[12 + i] = (_Float16)x3[i];
        }

        // ---- B fragments: 2 x ds_load_b128 each ----
        #pragma unroll
        for (int nt = 0; nt < 4; ++nt) {
            const int col = nt * 16 + (lane & 15);
            const _Float16* wp = &sWt[col * SWT + koff];
            v8h b0 = *(const v8h*)(wp);
            v8h b1 = *(const v8h*)(wp + 8);
            v16h bfrag;
            #pragma unroll
            for (int i = 0; i < 8; ++i) { bfrag[i] = b0[i]; bfrag[8 + i] = b1[i]; }
            acc[nt] = __builtin_amdgcn_wmma_f32_16x16x32_f16(
                false, a, false, bfrag, (short)0, acc[nt], false, false);
        }
    }

    // ---- pack C-layout accumulators through LDS, store coalesced b128 ----
    #pragma unroll
    for (int nt = 0; nt < 4; ++nt) {
        const int col = nt * 16 + (lane & 15);
        #pragma unroll
        for (int r = 0; r < 8; ++r)
            sOut[wave][(r + (hi ? 8 : 0)) * HS + col] = (_Float16)acc[nt][r];
    }
    __builtin_amdgcn_wave_barrier();
    {
        const _Float16* src = &sOut[wave][lane * 32];
        _Float16* dst = out + (size_t)(tile * 16) * HS + lane * 32;
        #pragma unroll
        for (int i = 0; i < 4; ++i)
            *(v8h*)(dst + i * 8) = *(const v8h*)(src + i * 8);
    }
}

// ---------------------------------------------------------------------------
// Kernel 2: flash attention. 4 waves/block, one 16-query tile each.
// Double-buffered staging: K via TDM, V via async engine, overlapped with the
// V-transpose + 8 WMMAs of the previous tile.
// ---------------------------------------------------------------------------
__global__ __launch_bounds__(128) void attn_kernel(
    const _Float16* __restrict__ q, const _Float16* __restrict__ k,
    const _Float16* __restrict__ v, float* __restrict__ out, int T)
{
    __shared__ __align__(16) _Float16 sK[2][BK * HS];    // 2 x 4KB
    __shared__ __align__(16) _Float16 sVraw[2][BK * HS]; // 2 x 4KB
    __shared__ __align__(16) _Float16 sVt[HS * SWT];     // V transposed (5KB)
    __shared__ float    sS[4][16][BK + 4];
    __shared__ __align__(16) _Float16 sP[4][16][BK];
    __shared__ float    sf[4][16];
    __shared__ float    sli[4][16];
    __shared__ float    sO[4][16 * HS];

    const int  tid   = threadIdx.x;
    const int  wave  = tid >> 5;
    const int  lane  = tid & 31;
    const bool hi    = lane >= 16;
    const int  qt    = blockIdx.x * 4 + wave;
    const int  b     = blockIdx.y;
    const int  qrow0 = qt * 16;
    const int  qmax  = qrow0 + 15;
    const size_t base = (size_t)b * (size_t)T * HS;

    const int aoff0 = hi ? 16 : 0;
    const int aoff1 = aoff0 + 8;
    const int koff  = hi ? 16 : 0;

    // Q fragments (A layout), straight from global f16
    const int arow = qrow0 + (lane & 15);
    v16h qa[2];
    #pragma unroll
    for (int c = 0; c < 2; ++c) {
        const _Float16* qr = q + base + (size_t)arow * HS + c * 32;
        v8h t0 = *(const v8h*)(qr + aoff0);
        v8h t1 = *(const v8h*)(qr + aoff1);
        #pragma unroll
        for (int i = 0; i < 8; ++i) { qa[c][i] = t0[i]; qa[c][8 + i] = t1[i]; }
    }

    v8f oacc[4] = {};
    float m = -1e30f, l = 0.0f;

    const int   nkt   = 2 * (blockIdx.x + 1);   // uniform loop bound per block
    const float scale = 0.03125f;               // 1024^-0.5

    auto stage = [&](int kt, int buf) {
        const int key0 = kt * BK;
        if (wave == 0)
            tdm_load_1d((unsigned)(unsigned long long)&sK[buf][0],
                        k + base + (size_t)key0 * HS, BK * HS);
        const char* vg = (const char*)(v + base + (size_t)key0 * HS) + tid * 32;
        const unsigned voff =
            (unsigned)(unsigned long long)&sVraw[buf][0] + tid * 32;
        async_g2l_b128(voff,      vg);
        async_g2l_b128(voff + 16, vg + 16);
    };

    stage(0, 0);   // prologue

    for (int kt = 0; kt < nkt; ++kt) {
        const int cur  = kt & 1;
        const int key0 = kt * BK;

        // wait for this tile's DMAs, publish, then kick off next tile's DMAs
        if (wave == 0) __builtin_amdgcn_s_wait_tensorcnt(0);
        wait_async0();
        __syncthreads();
        if (kt + 1 < nkt) stage(kt + 1, cur ^ 1);

        // ---- cooperative V transpose: sVraw[cur] -> sVt[dim][key] ----
        {
            const int dcol = tid & 63, rh = tid >> 6;
            v8h h0, h1;
            #pragma unroll
            for (int i = 0; i < 8; ++i) {
                h0[i] = sVraw[cur][(rh * 16 + i)     * HS + dcol];
                h1[i] = sVraw[cur][(rh * 16 + 8 + i) * HS + dcol];
            }
            *(v8h*)&sVt[dcol * SWT + rh * 16]     = h0;
            *(v8h*)&sVt[dcol * SWT + rh * 16 + 8] = h1;
        }
        __syncthreads();

        if (key0 <= qmax) {   // wave-uniform: EXEC all-ones for WMMA
            // ---- S = Q K^T over two 16-key subtiles ----
            #pragma unroll
            for (int n = 0; n < 2; ++n) {
                v8f s = {};
                const int klocal = n * 16 + (lane & 15);
                const int keycol = key0 + klocal;
                #pragma unroll
                for (int c = 0; c < 2; ++c) {
                    const _Float16* kr =
                        &sK[cur][klocal * HS + c * 32 + koff];
                    v8h t0 = *(const v8h*)(kr);
                    v8h t1 = *(const v8h*)(kr + 8);
                    v16h kb;
                    #pragma unroll
                    for (int i = 0; i < 8; ++i) { kb[i] = t0[i]; kb[8+i] = t1[i]; }
                    s = __builtin_amdgcn_wmma_f32_16x16x32_f16(
                        false, qa[c], false, kb, (short)0, s, false, false);
                }
                #pragma unroll
                for (int r = 0; r < 8; ++r) {
                    const int qq = qrow0 + r + (hi ? 8 : 0);
                    float sv = s[r] * scale;
                    if (keycol > qq) sv = -1e30f;
                    sS[wave][r + (hi ? 8 : 0)][klocal] = sv;
                }
            }
            __builtin_amdgcn_wave_barrier();

            // ---- online softmax: lanes 0..15 each own one query row ----
            if (lane < 16) {
                const int row = lane;
                float rmax = -1e30f;
                #pragma unroll
                for (int j = 0; j < BK; ++j) rmax = fmaxf(rmax, sS[wave][row][j]);
                const float mnew = fmaxf(m, rmax);
                const float corr = __expf(m - mnew);
                float ps = 0.0f;
                #pragma unroll
                for (int j = 0; j < BK; ++j) {
                    const float p = __expf(sS[wave][row][j] - mnew);
                    sP[wave][row][j] = (_Float16)p;
                    ps += p;
                }
                l = l * corr + ps;
                m = mnew;
                sf[wave][row] = corr;
            }
            __builtin_amdgcn_wave_barrier();

            // ---- rescale O, then O += P V ----
            float fr[8];
            #pragma unroll
            for (int r = 0; r < 8; ++r) fr[r] = sf[wave][r + (hi ? 8 : 0)];
            #pragma unroll
            for (int nt = 0; nt < 4; ++nt)
                #pragma unroll
                for (int r = 0; r < 8; ++r) oacc[nt][r] *= fr[r];

            v16h pa;   // P as A fragment: two contiguous 16B LDS runs
            {
                const _Float16* pp = &sP[wave][lane & 15][0];
                v8h p0 = *(const v8h*)(pp + aoff0);
                v8h p1 = *(const v8h*)(pp + aoff1);
                #pragma unroll
                for (int i = 0; i < 8; ++i) { pa[i] = p0[i]; pa[8 + i] = p1[i]; }
            }
            #pragma unroll
            for (int nt = 0; nt < 4; ++nt) {
                const int dim = nt * 16 + (lane & 15);
                const _Float16* vp = &sVt[dim * SWT + koff];
                v8h t0 = *(const v8h*)(vp);
                v8h t1 = *(const v8h*)(vp + 8);
                v16h vb;
                #pragma unroll
                for (int i = 0; i < 8; ++i) { vb[i] = t0[i]; vb[8 + i] = t1[i]; }
                oacc[nt] = __builtin_amdgcn_wmma_f32_16x16x32_f16(
                    false, pa, false, vb, (short)0, oacc[nt], false, false);
            }
        }
        __syncthreads();   // compute done before next iteration's staging reuse
    }

    // ---- finalize: O /= l, pack through LDS, coalesced b128 stores ----
    if (lane < 16) sli[wave][lane] = 1.0f / l;
    __builtin_amdgcn_wave_barrier();
    float li[8];
    #pragma unroll
    for (int r = 0; r < 8; ++r) li[r] = sli[wave][r + (hi ? 8 : 0)];
    #pragma unroll
    for (int nt = 0; nt < 4; ++nt) {
        const int col = nt * 16 + (lane & 15);
        #pragma unroll
        for (int r = 0; r < 8; ++r)
            sO[wave][(r + (hi ? 8 : 0)) * HS + col] = oacc[nt][r] * li[r];
    }
    __builtin_amdgcn_wave_barrier();
    {
        const float* src = &sO[wave][lane * 32];
        float* dst = out + base + (size_t)qrow0 * HS + lane * 32;
        #pragma unroll
        for (int i = 0; i < 8; ++i)
            *(v4f*)(dst + i * 4) = *(const v4f*)(src + i * 4);
    }
}

extern "C" void kernel_launch(void* const* d_in, const int* in_sizes, int n_in,
                              void* d_out, int out_size, void* d_ws, size_t ws_size,
                              hipStream_t stream) {
    const float* x  = (const float*)d_in[0];
    const float* Wq = (const float*)d_in[1];
    const float* Wk = (const float*)d_in[2];
    const float* Wv = (const float*)d_in[3];
    float* out = (float*)d_out;

    const int B = 4, T = 2048;
    const int BT = B * T;

    _Float16* qkv = (_Float16*)d_ws;              // 3 * BT * 64 f16 = 3 MB

    dim3 g1(BT / (16 * 4), 3);                    // 128 x 3 blocks, 4 waves each
    qkv_proj_kernel<<<g1, 128, 0, stream>>>(x, Wq, Wk, Wv, qkv, BT);

    const _Float16* qp = qkv;
    const _Float16* kp = qkv + (size_t)BT * HS;
    const _Float16* vp = qkv + 2 * (size_t)BT * HS;

    dim3 g2(T / 64, B);                           // 32 x 4 blocks, 4 waves each
    attn_kernel<<<g2, 128, 0, stream>>>(qp, kp, vp, out, T);
}
